// LaneATT_56727928046451
// MI455X (gfx1250) — compile-verified, matrix operand order checked by reading, not run on hardware
//
#include <hip/hip_runtime.h>
#include <hip/hip_bf16.h>

// ---------------- problem constants ----------------
constexpr int Bsz  = 16;
constexpr int CIN  = 512;
constexpr int CF   = 64;
constexpr int FH   = 11;
constexpr int FW   = 20;
constexpr int N    = 2784;
constexpr int NM1  = N - 1;        // 2783
constexpr int D    = CF * FH;      // 704
constexpr int CD2  = 2 * D;        // 1408
constexpr int HOUT = 75;           // 2 cls + 73 reg
constexpr int PROPW= 77;
constexpr long MROWS = (long)Bsz * N;   // 44544

// ---------------- WMMA types ----------------
typedef __attribute__((ext_vector_type(16))) __bf16 v16bf;
typedef __attribute__((ext_vector_type(8)))  float  v8f;

union FragU { uint4 u[2]; v16bf v; };
union Pack16 { uint4 u[2]; __hip_bfloat16 h[16]; };

// ---------------- CDNA5 async global->LDS helpers ----------------
// GLOBAL_LOAD_ASYNC_TO_LDS_B128: per-lane 16B copy straight to LDS,
// tracked by ASYNCcnt (no VGPR round trip, no wave stall at issue).
__device__ __forceinline__ void async_copy_b128(unsigned lds_off,
                                                const __hip_bfloat16* gp) {
  asm volatile("global_load_async_to_lds_b128 %0, %1, off"
               :: "v"(lds_off), "v"((unsigned long long)(uintptr_t)gp)
               : "memory");
}
__device__ __forceinline__ void wait_async0() {
  asm volatile("s_wait_asynccnt 0x0" ::: "memory");
}
// generic(shared) pointer -> 32-bit LDS byte offset (low 32 bits of aperture)
__device__ __forceinline__ unsigned lds_off32(const void* p) {
  return (unsigned)(unsigned long long)(uintptr_t)p;
}

// =====================================================================
// Kernel 1: 1x1 conv  feat[b,o,h,w] = sum_c x[b,c,h,w]*W[o,c] + bias[o]
// one block per (b,h); x[b,:,h,:] staged in LDS (512x20 f32 = 40 KB)
// =====================================================================
__global__ void conv1x1_kernel(const float* __restrict__ x,
                               const float* __restrict__ w,
                               const float* __restrict__ bias,
                               float* __restrict__ feat) {
  __shared__ float sx[CIN * FW];
  const int bh = blockIdx.x;
  const int b = bh / FH, h = bh % FH;
  const float* xp = x + (size_t)b * CIN * FH * FW + (size_t)h * FW;
  for (int i = threadIdx.x; i < CIN * FW; i += blockDim.x) {
    int c = i / FW, ww = i % FW;
    sx[i] = xp[(size_t)c * FH * FW + ww];
  }
  __syncthreads();
  for (int i = threadIdx.x; i < CF * FW; i += blockDim.x) {
    int o = i / FW, ww = i % FW;
    float acc = bias[o];
    const float* wr = w + (size_t)o * CIN;
    #pragma unroll 8
    for (int c = 0; c < CIN; ++c) acc += sx[c * FW + ww] * wr[c];
    feat[(((size_t)b * CF + o) * FH + h) * FW + ww] = acc;
  }
}

// =====================================================================
// Kernel 2: ROI gather + mask + f32->bf16, write into comb[:, D:2D]
// comb row layout: [att_feat(D) | anchor_feat(D)], ld = 1408 bf16
// =====================================================================
__global__ void gather_kernel(const float* __restrict__ feat,
                              const int* __restrict__ cut_xs,
                              const unsigned char* __restrict__ invalid,
                              __hip_bfloat16* __restrict__ comb) {
  long idx = (long)blockIdx.x * blockDim.x + threadIdx.x;
  const long total = MROWS * D;
  if (idx >= total) return;
  int d  = (int)(idx % D);
  long bn = idx / D;
  int n  = (int)(bn % N);
  int b  = (int)(bn / N);
  int o  = d / FH, h = d % FH;
  int cx = cut_xs[n * FH + h];
  float v = invalid[n * FH + h] ? 0.0f
          : feat[(((size_t)b * CF + o) * FH + h) * FW + cx];
  comb[(size_t)bn * CD2 + D + d] = __float2bfloat16(v);
}

// =====================================================================
// Kernel 3: f32 -> bf16 elementwise (attn_w)
// =====================================================================
__global__ void cvt_bf16_kernel(const float* __restrict__ src,
                                __hip_bfloat16* __restrict__ dst, long n) {
  long i = (long)blockIdx.x * blockDim.x + threadIdx.x;
  if (i < n) dst[i] = __float2bfloat16(src[i]);
}

// build fused head weights (75 x 1408 bf16) + bias (75 f32)
__global__ void build_head_kernel(const float* __restrict__ cls_w,
                                  const float* __restrict__ cls_b,
                                  const float* __restrict__ reg_w,
                                  const float* __restrict__ reg_b,
                                  __hip_bfloat16* __restrict__ wh,
                                  float* __restrict__ bh) {
  long i = (long)blockIdx.x * blockDim.x + threadIdx.x;
  if (i < HOUT) bh[i] = (i < 2) ? cls_b[i] : reg_b[i - 2];
  if (i >= (long)HOUT * CD2) return;
  int r = (int)(i / CD2), c = (int)(i % CD2);
  float v = (r < 2) ? cls_w[(size_t)r * CD2 + c] : reg_w[(size_t)(r - 2) * CD2 + c];
  wh[i] = __float2bfloat16(v);
}

// =====================================================================
// Kernel 4: tiled bf16 WMMA GEMM   C(MxNc) = A(MxK) * op(B) [+ bias]
//   TRANSB=1 : B is (Nc x K) row-major (C += A * B^T)
//   TRANSB=0 : B is (K x Nc) row-major
//   OUT_BF16 : C stored as bf16 (no bias), else f32 (+bias[n] if given)
// Block tile 128x128, KT=32; 8 waves (wave32); wave tile 32x64 (2x4 frags).
// Double-buffered LDS; in-bounds tiles staged via async global->LDS b128
// (ASYNCcnt) while WMMAs consume the other buffer; edge lanes ds-store
// zeros.  Fragments = two ds_load_b128 per lane (ISA 16-bit VGPR layout),
// LDS K-stride padded to 40 elements to avoid bank conflicts.
// =====================================================================
constexpr int BM = 128, BN = 128, KT = 32, LDK = 40;

template<bool TRANSB, bool OUT_BF16>
__global__ __launch_bounds__(256)
void gemm_bf16_kernel(const __hip_bfloat16* __restrict__ A, long sAb, int lda,
                      const __hip_bfloat16* __restrict__ Bm, long sBb, int ldb,
                      void* __restrict__ Cp, long sCb, int ldc,
                      const float* __restrict__ bias,
                      int Mrows, int Ncols, int K) {
  __shared__ __align__(16) __hip_bfloat16 sA[2][BM * LDK];
  __shared__ __align__(16) __hip_bfloat16 sB[2][BN * LDK];

  A  += (long)blockIdx.z * sAb;
  Bm += (long)blockIdx.z * sBb;
  char* Cb = (char*)Cp + (long)blockIdx.z * sCb * (OUT_BF16 ? 2 : 4);

  const int m0 = blockIdx.y * BM;
  const int n0 = blockIdx.x * BN;
  const int tid  = threadIdx.x;
  const int lane = tid & 31;
  const int wid  = tid >> 5;
  const int wm   = wid >> 1;          // 0..3  -> 32-row strip
  const int wn   = wid & 1;           // 0..1  -> 64-col strip
  const int l16  = lane & 15;
  const int half = lane >> 4;

  const uint4 z4 = {0u, 0u, 0u, 0u};
  const int ktiles = K / KT;

  // staging assignments (fixed per thread)
  const int arow  = tid >> 1;            // 0..127
  const int akseg = (tid & 1) * 16;      // 0 | 16
  const bool aok  = (m0 + arow) < Mrows;
  const unsigned aoff = lds_off32(&sA[0][arow * LDK + akseg]);

  auto stage = [&](int kt, int buf) {
    const unsigned bufA = (unsigned)(buf * BM * LDK * 2);
    // ---- A tile (BM x 32) ----
    {
      const __hip_bfloat16* gp = A + (size_t)(m0 + arow) * lda + kt * KT + akseg;
      if (aok) {
        async_copy_b128(aoff + bufA,      gp);
        async_copy_b128(aoff + bufA + 16, gp + 8);
      } else {
        *(uint4*)&sA[buf][arow * LDK + akseg]     = z4;
        *(uint4*)&sA[buf][arow * LDK + akseg + 8] = z4;
      }
    }
    // ---- B tile into sB[n][k] (K-contiguous) ----
    if (TRANSB) {
      const int nrow = tid >> 1;
      const int kseg = (tid & 1) * 16;
      const __hip_bfloat16* gp = Bm + (size_t)(n0 + nrow) * ldb + kt * KT + kseg;
      if (n0 + nrow < Ncols) {
        const unsigned boff = lds_off32(&sB[buf][nrow * LDK + kseg]);
        async_copy_b128(boff,      gp);
        async_copy_b128(boff + 16, gp + 8);
      } else {
        *(uint4*)&sB[buf][nrow * LDK + kseg]     = z4;
        *(uint4*)&sB[buf][nrow * LDK + kseg + 8] = z4;
      }
    } else {
      // B is K x Nc: transpose-on-store (sync path)
      const int k    = tid >> 3;          // 0..31
      const int nseg = (tid & 7) * 16;    // 0..112
      Pack16 t; t.u[0] = z4; t.u[1] = z4;
      if (n0 + nseg < Ncols) {            // Ncols % 16 == 0 in this mode
        const __hip_bfloat16* gp = Bm + (size_t)(kt * KT + k) * ldb + n0 + nseg;
        t.u[0] = *(const uint4*)gp;
        t.u[1] = *(const uint4*)(gp + 8);
      }
      #pragma unroll
      for (int i = 0; i < 16; ++i) sB[buf][(nseg + i) * LDK + k] = t.h[i];
    }
  };

  v8f acc[2][4] = {};

  stage(0, 0);
  for (int kt = 0; kt < ktiles; ++kt) {
    const int buf = kt & 1;
    wait_async0();          // staging of `buf` (issued last iter) complete
    __syncthreads();        // visible to all waves (also fences ds stores)
    if (kt + 1 < ktiles) stage(kt + 1, buf ^ 1);   // overlap with compute

    // ---- fragments: two contiguous 16B runs per lane (ISA 16-bit layout) ----
    FragU af[2], bf[4];
    #pragma unroll
    for (int i = 0; i < 2; ++i) {
      const __hip_bfloat16* p = &sA[buf][(wm * 32 + i * 16 + l16) * LDK + half * 8];
      af[i].u[0] = *(const uint4*)p;
      af[i].u[1] = *(const uint4*)(p + 16);
    }
    #pragma unroll
    for (int j = 0; j < 4; ++j) {
      const __hip_bfloat16* p = &sB[buf][(wn * 64 + j * 16 + l16) * LDK + half * 8];
      bf[j].u[0] = *(const uint4*)p;
      bf[j].u[1] = *(const uint4*)(p + 16);
    }
    #pragma unroll
    for (int i = 0; i < 2; ++i)
      #pragma unroll
      for (int j = 0; j < 4; ++j)
        acc[i][j] = __builtin_amdgcn_wmma_f32_16x16x32_bf16(
            false, af[i].v, false, bf[j].v, (short)0, acc[i][j], false, false);
  }

  // ---- store C (guarded), ISA 16x16 f32 C layout ----
  #pragma unroll
  for (int i = 0; i < 2; ++i) {
    const int mb = m0 + wm * 32 + i * 16 + half * 8;
    #pragma unroll
    for (int j = 0; j < 4; ++j) {
      const int n = n0 + wn * 64 + j * 16 + l16;
      if (n >= Ncols) continue;
      const float bv = (!OUT_BF16 && bias) ? bias[n] : 0.0f;
      #pragma unroll
      for (int r = 0; r < 8; ++r) {
        const int m = mb + r;
        if (m >= Mrows) continue;
        if (OUT_BF16)
          ((__hip_bfloat16*)Cb)[(size_t)m * ldc + n] = __float2bfloat16(acc[i][j][r]);
        else
          ((float*)Cb)[(size_t)m * ldc + n] = acc[i][j][r] + bv;
      }
    }
  }
}

// =====================================================================
// Kernel 5: row softmax over 2783 logits (stride-N rows written by GEMM1)
// then in-place expand to length-N row with 0 inserted at the diagonal;
// also emits the bf16 copy of M that feeds GEMM2.
// =====================================================================
__global__ __launch_bounds__(256)
void softmax_expand_kernel(float* __restrict__ Mout,
                           __hip_bfloat16* __restrict__ Mbf) {
  __shared__ float s[NM1];
  __shared__ float red[256];
  const long bn = blockIdx.x;
  const int  r  = (int)(bn % N);
  float* row = Mout + (size_t)bn * N;
  const int tid = threadIdx.x;

  float lmax = -3.0e38f;
  for (int k = tid; k < NM1; k += 256) { float v = row[k]; s[k] = v; lmax = fmaxf(lmax, v); }
  red[tid] = lmax; __syncthreads();
  for (int off = 128; off > 0; off >>= 1) {
    if (tid < off) red[tid] = fmaxf(red[tid], red[tid + off]);
    __syncthreads();
  }
  const float mx = red[0]; __syncthreads();

  float lsum = 0.0f;
  for (int k = tid; k < NM1; k += 256) { float e = __expf(s[k] - mx); s[k] = e; lsum += e; }
  red[tid] = lsum; __syncthreads();
  for (int off = 128; off > 0; off >>= 1) {
    if (tid < off) red[tid] += red[tid + off];
    __syncthreads();
  }
  const float inv = 1.0f / red[0];

  __hip_bfloat16* brow = Mbf + (size_t)bn * N;
  for (int c = tid; c < N; c += 256) {
    float v = 0.0f;
    if (c != r) v = s[c - (c > r)] * inv;
    row[c]  = v;
    brow[c] = __float2bfloat16(v);
  }
}

// =====================================================================
// Kernel 6: assemble props[b,n,0:77] = [cls(2) | anchors[2:4] | anchors[4:]+reg(73)]
// =====================================================================
__global__ void assemble_kernel(const float* __restrict__ headC,
                                const float* __restrict__ anchors,
                                float* __restrict__ props) {
  long idx = (long)blockIdx.x * blockDim.x + threadIdx.x;
  const long total = MROWS * PROPW;
  if (idx >= total) return;
  int j   = (int)(idx % PROPW);
  long bn = idx / PROPW;
  int n   = (int)(bn % N);
  float v;
  if (j < 2)      v = headC[(size_t)bn * HOUT + j];
  else if (j < 4) v = anchors[(size_t)n * PROPW + j];
  else            v = anchors[(size_t)n * PROPW + j] + headC[(size_t)bn * HOUT + (j - 2)];
  props[(size_t)bn * PROPW + j] = v;
}

// =====================================================================
// host launcher
// =====================================================================
extern "C" void kernel_launch(void* const* d_in, const int* in_sizes, int n_in,
                              void* d_out, int out_size, void* d_ws, size_t ws_size,
                              hipStream_t stream) {
  const float* x       = (const float*)d_in[0];
  const float* conv_w  = (const float*)d_in[1];
  const float* conv_b  = (const float*)d_in[2];
  const float* attn_w  = (const float*)d_in[3];
  const float* attn_b  = (const float*)d_in[4];
  const float* cls_w   = (const float*)d_in[5];
  const float* cls_b   = (const float*)d_in[6];
  const float* reg_w   = (const float*)d_in[7];
  const float* reg_b   = (const float*)d_in[8];
  const float* anchors = (const float*)d_in[9];
  const int*   cut_xs  = (const int*)d_in[10];
  const unsigned char* invalid = (const unsigned char*)d_in[11];

  float* props = (float*)d_out;                      // B*N*77
  float* Mout  = (float*)d_out + MROWS * PROPW;      // B*N*N (f32)

  // workspace carve-out (256B aligned)
  char* w = (char*)d_ws;
  auto alloc = [&](size_t bytes) { char* p = w; w += (bytes + 255) & ~(size_t)255; return p; };
  float*          feat    = (float*)          alloc((size_t)Bsz * CF * FH * FW * 4);     // 0.9 MB
  __hip_bfloat16* comb    = (__hip_bfloat16*) alloc((size_t)MROWS * CD2 * 2);            // 125 MB
  __hip_bfloat16* attn_wb = (__hip_bfloat16*) alloc((size_t)NM1 * D * 2);                // 3.9 MB
  __hip_bfloat16* Mbf     = (__hip_bfloat16*) alloc((size_t)MROWS * N * 2);              // 248 MB
  float*          headC   = (float*)          alloc((size_t)MROWS * HOUT * 4);           // 13.4 MB
  __hip_bfloat16* wh      = (__hip_bfloat16*) alloc((size_t)HOUT * CD2 * 2);             // 0.2 MB
  float*          bh      = (float*)          alloc((size_t)HOUT * 4);
  (void)ws_size; (void)in_sizes; (void)n_in; (void)out_size;

  // 1) 1x1 conv
  conv1x1_kernel<<<Bsz * FH, 256, 0, stream>>>(x, conv_w, conv_b, feat);

  // 2) gather -> comb[:, D:2D] (bf16 anchor features)
  {
    long total = MROWS * D;
    gather_kernel<<<(int)((total + 255) / 256), 256, 0, stream>>>(feat, cut_xs, invalid, comb);
  }

  // 3) weight conversions
  {
    long na = (long)NM1 * D;
    cvt_bf16_kernel<<<(int)((na + 255) / 256), 256, 0, stream>>>(attn_w, attn_wb, na);
    long nh = (long)HOUT * CD2;
    build_head_kernel<<<(int)((nh + 255) / 256), 256, 0, stream>>>(cls_w, cls_b, reg_w, reg_b, wh, bh);
  }

  // 4) GEMM1: scores = anchor_feat @ attn_w^T + attn_b
  //    written straight into the output M region with row stride N (diag slot
  //    left for the softmax/expand pass).  44544 x 2783, K = 704.
  {
    dim3 grid((NM1 + BN - 1) / BN, (int)(MROWS / BM), 1);  // 22 x 348
    gemm_bf16_kernel<true, false><<<grid, 256, 0, stream>>>(
        comb + D, 0, CD2, attn_wb, 0, D,
        (void*)Mout, 0, N, attn_b, (int)MROWS, NM1, D);
  }

  // 5) softmax + diagonal-insert expansion (f32 in d_out, bf16 copy in ws)
  softmax_expand_kernel<<<(int)MROWS, 256, 0, stream>>>(Mout, Mbf);

  // 6) GEMM2 (batched): att_feat = M @ anchor_feat -> comb[:, 0:D] (bf16)
  //    per batch: 2784 x 704, K = 2784
  {
    dim3 grid((D + BN - 1) / BN, (N + BM - 1) / BM, Bsz);  // 6 x 22 x 16
    gemm_bf16_kernel<false, true><<<grid, 256, 0, stream>>>(
        Mbf, (long)N * N, N,
        comb + D, (long)N * CD2, CD2,
        (void*)comb, (long)N * CD2, CD2,
        nullptr, N, D, N);
  }

  // 7) head GEMM: headC = comb @ [cls_w; reg_w]^T + bias   (44544 x 75, K=1408)
  {
    dim3 grid(1, (int)(MROWS / BM), 1);
    gemm_bf16_kernel<true, false><<<grid, 256, 0, stream>>>(
        comb, 0, CD2, wh, 0, CD2,
        (void*)headC, 0, HOUT, bh, (int)MROWS, HOUT, CD2);
  }

  // 8) assemble props
  {
    long total = MROWS * PROPW;
    assemble_kernel<<<(int)((total + 255) / 256), 256, 0, stream>>>(headC, anchors, props);
  }
}
